// T_prediction_20675972563727
// MI455X (gfx1250) — matrix operand order repaired and checked by reading, not compiled, not run on hardware
//
#include <hip/hip_runtime.h>
#include <math.h>

// ---------------------------------------------------------------------------
// Problem constants (from reference): B=4, D=512, N=2048, K=10, 45 pairs
// ---------------------------------------------------------------------------
#define B_    4
#define D_    512
#define N_    2048
#define TILE  16
#define KC    32            // bf16 WMMA K per instruction
#define NKC   (D_ / KC)     // 16 chunks
#define WAVES 4             // waves per block in kernel 1
#define KNN   10
#define NPAIR 45
#define GEPS  1e-6f
#define SHARP_ 20.0f
#define SCALE_ 441.94173824159216f   // 10000 / sqrt(512)

typedef __attribute__((ext_vector_type(16))) __bf16 v16bf;
typedef __attribute__((ext_vector_type(8)))  float  v8f;

// ---------------------------------------------------------------------------
// Kernel 1: fused  scores = srcE^T tgtE  (split-bf16 WMMA, 3 products)
//           + online softmax(10000*s/sqrt(D)) + argmax + corr_tgt = P @ tgt^T
// Grid: (N/16, B), block = 128 threads (4 waves). Each block owns 16 rows (n).
// A tile is staged in LDS *fragment-major*: aF[hi/lo][chunk][lane][16] holds
// each lane's 16-element WMMA A-fragment contiguously (32B, 32B-aligned), so
// the inner loop fetches it with vector DS loads instead of 16 scalar ones.
// ---------------------------------------------------------------------------
__global__ __launch_bounds__(128)
void scores_softmax_kernel(const float* __restrict__ srcE,
                           const float* __restrict__ tgtE,
                           const float* __restrict__ tgtP,
                           float* __restrict__ corres_out,   // (B,N) as float
                           float* __restrict__ corr_tgt)     // (B,N,3)
{
    __shared__ __attribute__((aligned(32))) __bf16 aF[2][NKC][32][16];
    __shared__ float  s_max[WAVES][TILE];
    __shared__ float  s_sum[WAVES][TILE];
    __shared__ float  s_acc[WAVES][TILE][3];
    __shared__ float  s_av [WAVES][TILE];
    __shared__ int    s_ai [WAVES][TILE];

    const int b    = blockIdx.y;
    const int n0   = blockIdx.x * TILE;
    const int tid  = threadIdx.x;
    const int wid  = tid >> 5;
    const int lane = tid & 31;

    // ---- stage A tile (16 rows x 512 K), split-bf16, fragment-major -------
    // ISA 16-bit A 16x32 layout: lane = row + 16*kin[3],
    //                            elem = kin[2:0] | (kin[4] << 3)
    const float* srcB = srcE + (size_t)b * D_ * N_;
    for (int i = tid; i < TILE * D_; i += 128) {
        int nl = i & (TILE - 1);
        int d  = i >> 4;
        float v  = srcB[(size_t)d * N_ + n0 + nl];
        __bf16 h = (__bf16)v;
        __bf16 l = (__bf16)(v - (float)h);
        int kc     = d >> 5;
        int kin    = d & 31;
        int lane_t = nl + ((kin >> 3) & 1) * 16;
        int e      = (kin & 7) | (((kin >> 4) & 1) << 3);
        aF[0][kc][lane_t][e] = h;
        aF[1][kc][lane_t][e] = l;
    }
    __syncthreads();

    const float* tgtB  = tgtE + (size_t)b * D_ * N_;
    const float* tgtPb = tgtP + (size_t)b * 3 * N_;
    const int half = lane >> 4;           // 0: lanes 0-15, 1: lanes 16-31
    const int l16  = lane & 15;
    const int kbB  = half ? 8 : 0;        // B frag K-base (ISA 16-bit B 32x16)

    // per-lane online softmax / argmax / weighted-point state (8 C rows)
    float rmax[8], rsum[8], ra0[8], ra1[8], ra2[8], rav[8];
    int   rai[8];
#pragma unroll
    for (int r = 0; r < 8; ++r) {
        rmax[r] = -1e30f; rsum[r] = 0.f;
        ra0[r] = ra1[r] = ra2[r] = 0.f;
        rav[r] = -1e30f; rai[r] = 0;
    }

    for (int mt = wid; mt < N_ / TILE; mt += WAVES) {
        const int m0  = mt * TILE;
        const int col = m0 + l16;
        v8f acc0 = {}; v8f acc1 = {}; v8f acc2 = {};

        for (int kc = 0; kc < NKC; ++kc) {
            const int k0 = kc * KC;
            if (kc + 1 < NKC)   // hint next K-chunk of B columns toward L1/L2
                __builtin_prefetch(&tgtB[(size_t)(k0 + KC + kbB) * N_ + col], 0, 1);

            // A fragments: one 32B vector LDS read each (fragment-major layout)
            v16bf ah = *(const v16bf*)&aF[0][kc][lane][0];
            v16bf al = *(const v16bf*)&aF[1][kc][lane][0];

            // B fragment from global/L2 (coalesced over l16): e -> K=kbB+(e&7)+16*(e>>3)
            v16bf bh, bl;
#pragma unroll
            for (int e = 0; e < 16; ++e) {
                float v  = tgtB[(size_t)(k0 + kbB + (e & 7) + ((e >> 3) << 4)) * N_ + col];
                __bf16 h = (__bf16)v;
                bh[e] = h;
                bl[e] = (__bf16)(v - (float)h);
            }
            // split-bf16 fp32-accurate product: hi*hi + hi*lo + lo*hi
            acc0 = __builtin_amdgcn_wmma_f32_16x16x32_bf16(false, ah, false, bh,
                                                           (short)0, acc0, false, false);
            acc1 = __builtin_amdgcn_wmma_f32_16x16x32_bf16(false, ah, false, bl,
                                                           (short)0, acc1, false, false);
            acc2 = __builtin_amdgcn_wmma_f32_16x16x32_bf16(false, al, false, bh,
                                                           (short)0, acc2, false, false);
        }

        // tgt point for this lane's column
        const int m = col;
        float t0 = tgtPb[m], t1 = tgtPb[N_ + m], t2 = tgtPb[2 * N_ + m];

#pragma unroll
        for (int r = 0; r < 8; ++r) {
            float s = (acc0[r] + acc1[r] + acc2[r]) * SCALE_;
            if (s > rav[r]) { rav[r] = s; rai[r] = m; }
            float nm = fmaxf(rmax[r], s);
            float c  = __expf(rmax[r] - nm);
            float e  = __expf(s - nm);
            rsum[r] = rsum[r] * c + e;
            ra0[r]  = ra0[r]  * c + e * t0;
            ra1[r]  = ra1[r]  * c + e * t1;
            ra2[r]  = ra2[r]  * c + e * t2;
            rmax[r] = nm;
        }
    }

    // ---- merge softmax states across the 16 lanes of each half-wave -------
#pragma unroll
    for (int off = 8; off >= 1; off >>= 1) {
#pragma unroll
        for (int r = 0; r < 8; ++r) {
            float om = __shfl_xor(rmax[r], off);
            float os = __shfl_xor(rsum[r], off);
            float o0 = __shfl_xor(ra0[r],  off);
            float o1 = __shfl_xor(ra1[r],  off);
            float o2 = __shfl_xor(ra2[r],  off);
            float ov = __shfl_xor(rav[r],  off);
            int   oi = __shfl_xor(rai[r],  off);
            float nm = fmaxf(rmax[r], om);
            float ca = __expf(rmax[r] - nm), cb = __expf(om - nm);
            rsum[r] = rsum[r] * ca + os * cb;
            ra0[r]  = ra0[r]  * ca + o0 * cb;
            ra1[r]  = ra1[r]  * ca + o1 * cb;
            ra2[r]  = ra2[r]  * ca + o2 * cb;
            rmax[r] = nm;
            if (ov > rav[r] || (ov == rav[r] && oi < rai[r])) { rav[r] = ov; rai[r] = oi; }
        }
    }

    // one writer lane per row -> LDS
#pragma unroll
    for (int r = 0; r < 8; ++r) {
        if (l16 == r) {
            int row = r + 8 * half;
            s_max[wid][row]    = rmax[r];
            s_sum[wid][row]    = rsum[r];
            s_acc[wid][row][0] = ra0[r];
            s_acc[wid][row][1] = ra1[r];
            s_acc[wid][row][2] = ra2[r];
            s_av [wid][row]    = rav[r];
            s_ai [wid][row]    = rai[r];
        }
    }
    __syncthreads();

    // ---- merge across waves, finalize row n0+row --------------------------
    if (tid < TILE) {
        int row = tid;
        float M = s_max[0][row];
        for (int w = 1; w < WAVES; ++w) M = fmaxf(M, s_max[w][row]);
        float S = 0.f, A0 = 0.f, A1 = 0.f, A2 = 0.f, av = -1e30f;
        int ai = 0;
        for (int w = 0; w < WAVES; ++w) {
            float c = __expf(s_max[w][row] - M);
            S  += s_sum[w][row] * c;
            A0 += s_acc[w][row][0] * c;
            A1 += s_acc[w][row][1] * c;
            A2 += s_acc[w][row][2] * c;
            float v = s_av[w][row]; int ii = s_ai[w][row];
            if (v > av || (v == av && ii < ai)) { av = v; ai = ii; }
        }
        int n = n0 + row;
        corres_out[(size_t)b * N_ + n] = (float)ai;
        float inv = 1.f / S;
        float* ct = corr_tgt + ((size_t)b * N_ + n) * 3;
        ct[0] = A0 * inv; ct[1] = A1 * inv; ct[2] = A2 * inv;
    }
}

// ---------------------------------------------------------------------------
// Kernel 2: GFM per-point loss (KNN k=10 over src, 45 triangle pairs) + per-
// batch min via uint atomicMin (L >= 0 so the bit pattern is order-preserving)
// ---------------------------------------------------------------------------
__global__ void init_min_kernel(unsigned int* minbits)
{
    if (threadIdx.x < B_) minbits[threadIdx.x] = 0x7f800000u; // +inf
}

__global__ __launch_bounds__(128)
void gfm_loss_kernel(const float* __restrict__ srcP,
                     const float* __restrict__ corr_tgt,
                     float* __restrict__ Lout,
                     unsigned int* __restrict__ minbits)
{
    int gid = blockIdx.x * blockDim.x + threadIdx.x;
    if (gid >= B_ * N_) return;
    int b = gid / N_, n = gid - b * N_;
    const float* sp = srcP + (size_t)b * 3 * N_;
    float p0x = sp[n], p0y = sp[N_ + n], p0z = sp[2 * N_ + n];

    // --- k nearest (ascending d2 == descending exp(-d2/2); ties keep low idx)
    float bd[KNN]; int bi[KNN];
#pragma unroll
    for (int j = 0; j < KNN; ++j) { bd[j] = 1e30f; bi[j] = 0; }
    for (int m = 0; m < N_; ++m) {
        float dx = sp[m] - p0x, dy = sp[N_ + m] - p0y, dz = sp[2 * N_ + m] - p0z;
        float d2 = dx * dx + dy * dy + dz * dz;
        if (d2 < bd[KNN - 1]) {
            int j = KNN - 1;
            while (j > 0 && bd[j - 1] > d2) { bd[j] = bd[j - 1]; bi[j] = bi[j - 1]; --j; }
            bd[j] = d2; bi[j] = m;
        }
    }

    float nsx[KNN], nsy[KNN], nsz[KNN], ntx[KNN], nty[KNN], ntz[KNN];
#pragma unroll
    for (int j = 0; j < KNN; ++j) {
        int m = bi[j];
        nsx[j] = sp[m]; nsy[j] = sp[N_ + m]; nsz[j] = sp[2 * N_ + m];
        const float* ct = corr_tgt + ((size_t)b * N_ + m) * 3;
        ntx[j] = ct[0]; nty[j] = ct[1]; ntz[j] = ct[2];
    }
    const float* ct0 = corr_tgt + ((size_t)b * N_ + n) * 3;
    float q0x = ct0[0], q0y = ct0[1], q0z = ct0[2];

    // --- 45 pairs: ls=[0,0,as], lt=[eps,eps,at+eps]
    float loss[NPAIR], reg[NPAIR];
    int p = 0;
    for (int i = 0; i < KNN; ++i) {
        for (int j = i + 1; j < KNN; ++j) {
            float e1x = nsx[i] - p0x, e1y = nsy[i] - p0y, e1z = nsz[i] - p0z;
            float e2x = nsx[j] - p0x, e2y = nsy[j] - p0y, e2z = nsz[j] - p0z;
            float cx = e1y * e2z - e1z * e2y;
            float cy = e1z * e2x - e1x * e2z;
            float cz = e1x * e2y - e1y * e2x;
            float as_ = 0.5f * sqrtf(cx * cx + cy * cy + cz * cz);

            float f1x = ntx[i] - q0x, f1y = nty[i] - q0y, f1z = ntz[i] - q0z;
            float f2x = ntx[j] - q0x, f2y = nty[j] - q0y, f2z = ntz[j] - q0z;
            float gx = f1y * f2z - f1z * f2y;
            float gy = f1z * f2x - f1x * f2z;
            float gz = f1x * f2y - f1y * f2x;
            float at_ = 0.5f * sqrtf(gx * gx + gy * gy + gz * gz);

            float dm = as_ - at_ - GEPS;
            float sm = as_ + at_ + GEPS;
            float num = 2.f * GEPS * GEPS + dm * dm;
            float den = 2.f * GEPS * GEPS + sm * sm;
            loss[p] = num / den;
            reg[p]  = sqrtf(num);
            ++p;
        }
    }
    // sort loss ascending; reg stays in original pair order (matches reference)
    for (int i = 1; i < NPAIR; ++i) {
        float key = loss[i]; int j = i - 1;
        while (j >= 0 && loss[j] > key) { loss[j + 1] = loss[j]; --j; }
        loss[j + 1] = key;
    }
    float comb[NPAIR], tmp[NPAIR];
    for (int i = 0; i < NPAIR; ++i) { comb[i] = loss[i] + 0.1f * reg[i]; tmp[i] = comb[i]; }
    for (int i = 1; i < NPAIR; ++i) {                 // median of 45 -> tmp[22]
        float key = tmp[i]; int j = i - 1;
        while (j >= 0 && tmp[j] > key) { tmp[j + 1] = tmp[j]; --j; }
        tmp[j + 1] = key;
    }
    float med = tmp[22];
    float L = 0.f;
    for (int i = 0; i < KNN; ++i) {
        float v = comb[i] > 3.f * med ? 0.f : comb[i];
        L += sqrtf(v + GEPS);
    }
    L *= (1.f / KNN);
    Lout[gid] = L;
    atomicMin(&minbits[b], __float_as_uint(L));
}

// ---------------------------------------------------------------------------
// Kernel 3: weight = (2*sigmoid(-SHARP*(L - minL)) > 0.5) ? 1 : 0
// ---------------------------------------------------------------------------
__global__ __launch_bounds__(256)
void weight_kernel(const float* __restrict__ Lin,
                   const unsigned int* __restrict__ minbits,
                   float* __restrict__ wout)
{
    int gid = blockIdx.x * blockDim.x + threadIdx.x;
    if (gid >= B_ * N_) return;
    int b = gid / N_;
    float l = Lin[gid] - __uint_as_float(minbits[b]);
    float w = 2.f / (1.f + __expf(SHARP_ * l));
    wout[gid] = (w > 0.5f) ? 1.f : 0.f;
}

// ---------------------------------------------------------------------------
// Kernel 4: weighted Procrustes per batch (block reduce + 3x3 Jacobi SVD)
// ---------------------------------------------------------------------------
__device__ __forceinline__ float blockReduceSum(float v, float* red)
{
    int tid = threadIdx.x;
    red[tid] = v; __syncthreads();
    for (int s = 128; s > 0; s >>= 1) {
        if (tid < s) red[tid] += red[tid + s];
        __syncthreads();
    }
    float r = red[0]; __syncthreads();
    return r;
}

__global__ __launch_bounds__(256)
void procrustes_kernel(const float* __restrict__ srcP,
                       const float* __restrict__ tgtP,
                       const float* __restrict__ corres,
                       const float* __restrict__ weight,
                       float* __restrict__ Rout,
                       float* __restrict__ Tout)
{
    __shared__ float red[256];
    const int b = blockIdx.x, tid = threadIdx.x;
    const float* sp = srcP + (size_t)b * 3 * N_;
    const float* tp = tgtP + (size_t)b * 3 * N_;
    const float* wv = weight + (size_t)b * N_;
    const float* cr = corres + (size_t)b * N_;

    float lw = 0, lx0 = 0, lx1 = 0, lx2 = 0, ly0 = 0, ly1 = 0, ly2 = 0;
    for (int n = tid; n < N_; n += 256) {
        float w = wv[n];
        int   m = (int)cr[n];
        lw  += w;
        lx0 += w * sp[n];       lx1 += w * sp[N_ + n];   lx2 += w * sp[2 * N_ + n];
        ly0 += w * tp[m];       ly1 += w * tp[N_ + m];   ly2 += w * tp[2 * N_ + m];
    }
    float W1  = blockReduceSum(lw,  red);
    float SX0 = blockReduceSum(lx0, red), SX1 = blockReduceSum(lx1, red), SX2 = blockReduceSum(lx2, red);
    float SY0 = blockReduceSum(ly0, red), SY1 = blockReduceSum(ly1, red), SY2 = blockReduceSum(ly2, red);
    float inv = 1.f / (W1 + 1e-7f);
    float mux0 = SX0 * inv, mux1 = SX1 * inv, mux2 = SX2 * inv;
    float muy0 = SY0 * inv, muy1 = SY1 * inv, muy2 = SY2 * inv;

    float m00 = 0, m01 = 0, m02 = 0, m10 = 0, m11 = 0, m12 = 0, m20 = 0, m21 = 0, m22 = 0;
    for (int n = tid; n < N_; n += 256) {
        float w = wv[n];
        if (w != 0.f) {
            int m = (int)cr[n];
            float x0 = sp[n] - mux0, x1 = sp[N_ + n] - mux1, x2 = sp[2 * N_ + n] - mux2;
            float y0 = tp[m] - muy0, y1 = tp[N_ + m] - muy1, y2 = tp[2 * N_ + m] - muy2;
            m00 += w * y0 * x0; m01 += w * y0 * x1; m02 += w * y0 * x2;
            m10 += w * y1 * x0; m11 += w * y1 * x1; m12 += w * y1 * x2;
            m20 += w * y2 * x0; m21 += w * y2 * x1; m22 += w * y2 * x2;
        }
    }
    float M_[3][3];
    M_[0][0] = blockReduceSum(m00, red) * inv; M_[0][1] = blockReduceSum(m01, red) * inv;
    M_[0][2] = blockReduceSum(m02, red) * inv; M_[1][0] = blockReduceSum(m10, red) * inv;
    M_[1][1] = blockReduceSum(m11, red) * inv; M_[1][2] = blockReduceSum(m12, red) * inv;
    M_[2][0] = blockReduceSum(m20, red) * inv; M_[2][1] = blockReduceSum(m21, red) * inv;
    M_[2][2] = blockReduceSum(m22, red) * inv;

    if (tid == 0) {
        // S = M^T M ; Jacobi eigendecomposition -> V, then U = M V / sigma
        float S[3][3], V[3][3];
        for (int i = 0; i < 3; ++i)
            for (int j = 0; j < 3; ++j) {
                float a = 0.f;
                for (int k = 0; k < 3; ++k) a += M_[k][i] * M_[k][j];
                S[i][j] = a;
                V[i][j] = (i == j) ? 1.f : 0.f;
            }
        for (int sweep = 0; sweep < 24; ++sweep) {
            const int PQ[3][2] = {{0, 1}, {0, 2}, {1, 2}};
            for (int r = 0; r < 3; ++r) {
                int p = PQ[r][0], q = PQ[r][1];
                float apq = S[p][q];
                if (fabsf(apq) < 1e-20f) continue;
                float tau = (S[q][q] - S[p][p]) / (2.f * apq);
                float t = ((tau >= 0.f) ? 1.f : -1.f) / (fabsf(tau) + sqrtf(1.f + tau * tau));
                float c = 1.f / sqrtf(1.f + t * t), s = t * c;
                float app = S[p][p], aqq = S[q][q];
                S[p][p] = app - t * apq;
                S[q][q] = aqq + t * apq;
                S[p][q] = S[q][p] = 0.f;
                int k = 3 - p - q;   // the remaining index
                float skp = S[k][p], skq = S[k][q];
                S[k][p] = S[p][k] = c * skp - s * skq;
                S[k][q] = S[q][k] = s * skp + c * skq;
                for (int kk = 0; kk < 3; ++kk) {
                    float vkp = V[kk][p], vkq = V[kk][q];
                    V[kk][p] = c * vkp - s * vkq;
                    V[kk][q] = s * vkp + c * vkq;
                }
            }
        }
        float ev[3] = { S[0][0], S[1][1], S[2][2] };
        // sort eigenvalues descending with V column swaps
        for (int i = 0; i < 2; ++i)
            for (int j = 0; j < 2 - i; ++j)
                if (ev[j] < ev[j + 1]) {
                    float t = ev[j]; ev[j] = ev[j + 1]; ev[j + 1] = t;
                    for (int k = 0; k < 3; ++k) {
                        float tv = V[k][j]; V[k][j] = V[k][j + 1]; V[k][j + 1] = tv;
                    }
                }
        float U[3][3];
        for (int i = 0; i < 2; ++i) {
            float sg = sqrtf(fmaxf(ev[i], 0.f));
            float isg = (sg > 1e-20f) ? 1.f / sg : 0.f;
            for (int r = 0; r < 3; ++r)
                U[r][i] = (M_[r][0] * V[0][i] + M_[r][1] * V[1][i] + M_[r][2] * V[2][i]) * isg;
        }
        float sg2 = sqrtf(fmaxf(ev[2], 0.f));
        if (sg2 > 1e-12f * (sqrtf(fmaxf(ev[0], 0.f)) + 1e-30f)) {
            float isg = 1.f / sg2;
            for (int r = 0; r < 3; ++r)
                U[r][2] = (M_[r][0] * V[0][2] + M_[r][1] * V[1][2] + M_[r][2] * V[2][2]) * isg;
        } else {  // degenerate: complete the basis
            U[0][2] = U[1][0] * U[2][1] - U[2][0] * U[1][1];
            U[1][2] = U[2][0] * U[0][1] - U[0][0] * U[2][1];
            U[2][2] = U[0][0] * U[1][1] - U[1][0] * U[0][1];
        }
        float detU = U[0][0] * (U[1][1] * U[2][2] - U[1][2] * U[2][1])
                   - U[0][1] * (U[1][0] * U[2][2] - U[1][2] * U[2][0])
                   + U[0][2] * (U[1][0] * U[2][1] - U[1][1] * U[2][0]);
        float detV = V[0][0] * (V[1][1] * V[2][2] - V[1][2] * V[2][1])
                   - V[0][1] * (V[1][0] * V[2][2] - V[1][2] * V[2][0])
                   + V[0][2] * (V[1][0] * V[2][1] - V[1][1] * V[2][0]);
        float d3 = (detU * detV < 0.f) ? -1.f : 1.f;
        float R[3][3];
        for (int i = 0; i < 3; ++i)
            for (int j = 0; j < 3; ++j)
                R[i][j] = U[i][0] * V[j][0] + U[i][1] * V[j][1] + d3 * U[i][2] * V[j][2];
        float* Rb = Rout + b * 9;
        for (int i = 0; i < 3; ++i)
            for (int j = 0; j < 3; ++j) Rb[i * 3 + j] = R[i][j];
        float mux[3] = { mux0, mux1, mux2 }, muy[3] = { muy0, muy1, muy2 };
        float* Tb = Tout + b * 3;
        for (int i = 0; i < 3; ++i)
            Tb[i] = muy[i] - (R[i][0] * mux[0] + R[i][1] * mux[1] + R[i][2] * mux[2]);
    }
}

// ---------------------------------------------------------------------------
// Launch: d_out = [ R(36) | T(12) | corres(8192) | weight(8192) ]  (floats)
// d_ws   = [ corr_tgt(24576 f) | L(8192 f) | minbits(4 u32) ]
// ---------------------------------------------------------------------------
extern "C" void kernel_launch(void* const* d_in, const int* in_sizes, int n_in,
                              void* d_out, int out_size, void* d_ws, size_t ws_size,
                              hipStream_t stream)
{
    (void)in_sizes; (void)n_in; (void)out_size; (void)ws_size;
    const float* srcE = (const float*)d_in[0];
    const float* tgtE = (const float*)d_in[1];
    const float* srcP = (const float*)d_in[2];
    const float* tgtP = (const float*)d_in[3];

    float* out        = (float*)d_out;
    float* R_out      = out;
    float* T_out      = out + B_ * 9;
    float* corres_out = out + B_ * 9 + B_ * 3;
    float* weight_out = corres_out + B_ * N_;

    float* ws        = (float*)d_ws;
    float* corr_tgt  = ws;
    float* Lbuf      = ws + (size_t)B_ * N_ * 3;
    unsigned int* mn = (unsigned int*)(Lbuf + (size_t)B_ * N_);

    dim3 g1(N_ / TILE, B_);
    scores_softmax_kernel<<<g1, 128, 0, stream>>>(srcE, tgtE, tgtP, corres_out, corr_tgt);
    init_min_kernel<<<1, 32, 0, stream>>>(mn);
    gfm_loss_kernel<<<(B_ * N_ + 127) / 128, 128, 0, stream>>>(srcP, corr_tgt, Lbuf, mn);
    weight_kernel<<<(B_ * N_ + 255) / 256, 256, 0, stream>>>(Lbuf, mn, weight_out);
    procrustes_kernel<<<B_, 256, 0, stream>>>(srcP, tgtP, corres_out, weight_out, R_out, T_out);
}
